// CNN_VQ_CIFAR_50079318671808
// MI455X (gfx1250) — matrix operand order repaired and checked by reading, not compiled, not run on hardware
//
#include <hip/hip_runtime.h>

typedef __attribute__((ext_vector_type(16))) _Float16 v16h;
typedef __attribute__((ext_vector_type(8)))  _Float16 v8h;
typedef __attribute__((ext_vector_type(4)))  _Float16 v4h;
typedef __attribute__((ext_vector_type(8)))  float    v8f;
typedef __attribute__((ext_vector_type(4)))  float    v4f;
typedef __attribute__((ext_vector_type(4)))  unsigned int v4u;
typedef __attribute__((ext_vector_type(8)))  int      v8i;
typedef __attribute__((ext_vector_type(4)))  int      v4i;

// ---------------------------------------------------------------------------
// WMMA + cross-lane helpers (CDNA5 wave32)
// ---------------------------------------------------------------------------
__device__ __forceinline__ v8f wmma32(v16h a, v16h b, v8f c) {
  return __builtin_amdgcn_wmma_f32_16x16x32_f16(false, a, false, b, (short)0, c,
                                                false, false);
}

// XOR lane shuffle via ds_swizzle (group-of-32 mode: and=0x1f, or=0, xor=M)
template <int M>
__device__ __forceinline__ float swzf(float v) {
  return __int_as_float(
      __builtin_amdgcn_ds_swizzle(__float_as_int(v), (M << 10) | 0x1f));
}
template <int M>
__device__ __forceinline__ int swzi(int v) {
  return __builtin_amdgcn_ds_swizzle(v, (M << 10) | 0x1f);
}
template <int M>
__device__ __forceinline__ void argmin_step(float& d, int& i) {
  float d2 = swzf<M>(d);
  int   i2 = swzi<M>(i);
  if (d2 < d || (d2 == d && i2 < i)) { d = d2; i = i2; }
}

union Frag { v16h v; v8h h[2]; };

// A fragment (16x32 f16): one contiguous run of 16 halves per lane.
__device__ __forceinline__ v16h load_afrag(const _Float16* base, int lane) {
  const v8h* p = (const v8h*)(base + ((lane < 16) ? 0 : 16));
  Frag f; f.h[0] = p[0]; f.h[1] = p[1];
  return f.v;
}
// B fragment (32x16 f16): one contiguous run of 16 halves per lane.
__device__ __forceinline__ v16h load_bfrag(const _Float16* base, int lane) {
  const v8h* p = (const v8h*)(base + ((lane < 16) ? 0 : 16));
  Frag f; f.h[0] = p[0]; f.h[1] = p[1];
  return f.v;
}
__device__ __forceinline__ int bfk(int lane, int j) {
  return ((lane < 16) ? 0 : 16) + j;
}
__device__ __forceinline__ int afk(int lane, int j) {
  return ((lane < 16) ? 0 : 16) + j;
}

// ---------------------------------------------------------------------------
// Workspace layout (bytes, all 16B aligned)
// ---------------------------------------------------------------------------
constexpr size_t OFF_A1   = 0;                         // f16 [2048][64][15][15]
constexpr size_t OFF_A2   = OFF_A1 + 29491200ull * 2;  // f16 [2048][32][6][6]
constexpr size_t OFF_A3   = OFF_A2 + 2359296ull * 2;   // f16 [2048][32][16]
constexpr size_t OFF_BL   = OFF_A3 + 1048576ull * 2;   // f16 [2048][512]
constexpr size_t OFF_W1   = OFF_BL + 1048576ull * 2;   // f16 [64][32]  (pad 27->32)
constexpr size_t OFF_W2   = OFF_W1 + 4096;             // f16 [32][576]
constexpr size_t OFF_W3   = OFF_W2 + 36864;            // f16 [32][288]
constexpr size_t OFF_CB   = OFF_W3 + 18432;            // f16 [112][32] (pad 100->112)
constexpr size_t OFF_F1   = OFF_CB + 7168;             // f16 [64][512]
constexpr size_t OFF_F2   = OFF_F1 + 65536;            // f16 [16][64]  (pad 10->16)
constexpr size_t OFF_BN1A = OFF_F2 + 2048;             // f32 [64]
constexpr size_t OFF_BN1B = OFF_BN1A + 256;
constexpr size_t OFF_BN2A = OFF_BN1B + 256;            // f32 [32]
constexpr size_t OFF_BN2B = OFF_BN2A + 128;
constexpr size_t OFF_BN3A = OFF_BN2B + 128;
constexpr size_t OFF_BN3B = OFF_BN3A + 128;
constexpr size_t OFF_CC   = OFF_BN3B + 128;            // f32 [112] (pad = +huge)
constexpr size_t OFF_DUMP = OFF_CC + 512;              // 64KB store-sink scratch

// ---------------------------------------------------------------------------
// Weight f16 conversion with zero-padding + codebook norms + loss zeroing
// ---------------------------------------------------------------------------
__global__ __launch_bounds__(256) void k_prep(
    const float* __restrict__ w1, const float* __restrict__ w2,
    const float* __restrict__ w3, const float* __restrict__ cb,
    const float* __restrict__ f1, const float* __restrict__ f2,
    _Float16* __restrict__ w1p, _Float16* __restrict__ w2h,
    _Float16* __restrict__ w3h, _Float16* __restrict__ cbp,
    _Float16* __restrict__ f1h, _Float16* __restrict__ f2p,
    float* __restrict__ ccsq, float* __restrict__ loss) {
  int i = blockIdx.x * 256 + threadIdx.x;
  if (i == 0) *loss = 0.f;
  if (i < 2048) {                       // conv1_w [64][27] -> [64][32] zero-pad
    int row = i >> 5, k = i & 31;
    w1p[i] = (k < 27) ? (_Float16)w1[row * 27 + k] : (_Float16)0.f;
  }
  if (i < 18432) w2h[i] = (_Float16)w2[i];
  if (i < 9216)  w3h[i] = (_Float16)w3[i];
  if (i < 3584)  cbp[i] = (i < 3200) ? (_Float16)cb[i] : (_Float16)0.f;
  if (i < 32768) f1h[i] = (_Float16)f1[i];
  if (i < 1024)  f2p[i] = (i < 640) ? (_Float16)f2[i] : (_Float16)0.f;
  if (i < 112) {
    float s = 3.0e38f;                  // padded codes: never win argmin
    if (i < 100) {
      s = 0.f;
      for (int k = 0; k < 32; ++k) { float v = cb[i * 32 + k]; s = fmaf(v, v, s); }
    }
    ccsq[i] = s;
  }
}

// ---------------------------------------------------------------------------
// conv1 (3x3 VALID, 3->64ch) + maxpool2 + relu. Implicit GEMM, K=27 pad 32.
// ---------------------------------------------------------------------------
__global__ __launch_bounds__(256) void k_conv1(const float* __restrict__ x,
    const _Float16* __restrict__ w1p, const float* __restrict__ bias1,
    _Float16* __restrict__ a1h, _Float16* __restrict__ dump) {
  int task = blockIdx.x * 8 + (threadIdx.x >> 5);  // 0..61439 = 2048*15*2
  int lane = threadIdx.x & 31;
  int nl   = lane & 15;
  int ct   = task & 1;
  int pr   = (task >> 1) % 15;
  int n    = task / 30;
  int C    = ct * 16 + nl;
  int Cc   = (C < 30) ? C : 0;          // address clamp; col discarded later
  int R0   = 2 * pr;
  const float* xb = x + (size_t)n * 3072;

  v16h b0, b1v;
  #pragma unroll
  for (int j = 0; j < 16; ++j) {
    int k  = bfk(lane, j);
    int kk = (k < 27) ? k : 0;          // address clamp; A rows k>=27 are zero
    int ci = kk / 9, kh = (kk % 9) / 3, kw = kk % 3;
    const float* p = xb + ci * 1024 + (R0 + kh) * 32 + Cc + kw;
    b0[j]  = (_Float16)p[0];
    b1v[j] = (_Float16)p[32];
  }

  bool valid = ((nl & 1) == 0) && (C + 1) < 30;
  _Float16* op = valid ? (a1h + (size_t)n * 14400 + pr * 15 + (C >> 1)) : dump;

  #pragma unroll
  for (int mt = 0; mt < 4; ++mt) {
    v16h a = load_afrag(w1p + (mt * 16 + nl) * 32, lane);
    v8f c0 = {}, c1 = {};
    c0 = wmma32(a, b0, c0);
    c1 = wmma32(a, b1v, c1);
    #pragma unroll
    for (int r = 0; r < 8; ++r) {
      int mg  = mt * 16 + r + ((lane >= 16) ? 8 : 0);
      float v = fmaxf(c0[r], c1[r]) + bias1[mg];   // row-pool + bias
      v = fmaxf(v, swzf<1>(v));                    // col-pool
      op[mg * 225] = (_Float16)fmaxf(v, 0.f);      // relu, unconditional store
    }
  }
}

// ---------------------------------------------------------------------------
// Per-channel BN stats -> fused affine (A = rstd*scale, B = bias - mean*A)
// ---------------------------------------------------------------------------
template <int C, int S>
__global__ __launch_bounds__(256) void k_bnstats(const _Float16* __restrict__ act,
    const float* __restrict__ scale, const float* __restrict__ bias,
    float* __restrict__ outA, float* __restrict__ outB) {
  __shared__ float ssum[256], ssq[256];
  int c = blockIdx.x, t = threadIdx.x;
  float s = 0.f, q = 0.f;
  for (int i = t; i < 2048 * S; i += 256) {
    int nimg = i / S, p = i - nimg * S;
    float v = (float)act[(size_t)nimg * C * S + c * S + p];
    s += v; q = fmaf(v, v, q);
  }
  ssum[t] = s; ssq[t] = q;
  __syncthreads();
  for (int o = 128; o > 0; o >>= 1) {
    if (t < o) { ssum[t] += ssum[t + o]; ssq[t] += ssq[t + o]; }
    __syncthreads();
  }
  if (t == 0) {
    float inv  = 1.f / (2048.f * (float)S);
    float mean = ssum[0] * inv;
    float var  = ssq[0] * inv - mean * mean;
    float A    = rsqrtf(var + 1e-5f) * scale[c];
    outA[c] = A;
    outB[c] = bias[c] - mean * A;
  }
}

// In-place BN application, one block per image ([C][S] per block).
template <int C, int S>
__global__ __launch_bounds__(256) void k_bnapply(_Float16* __restrict__ act,
    const float* __restrict__ A, const float* __restrict__ B) {
  _Float16* p = act + (size_t)blockIdx.x * C * S;
  for (int i = threadIdx.x; i < C * S; i += 256) {
    int c = i / S;
    p[i] = (_Float16)fmaf((float)p[i], A[c], B[c]);
  }
}

// BN apply + coalesced f32 mirror (produces the h output tensor).
template <int C, int S>
__global__ __launch_bounds__(256) void k_bnapply_out(_Float16* __restrict__ act,
    const float* __restrict__ A, const float* __restrict__ B,
    float* __restrict__ outp) {
  _Float16* p = act + (size_t)blockIdx.x * C * S;
  float*    o = outp + (size_t)blockIdx.x * C * S;
  for (int i = threadIdx.x; i < C * S; i += 256) {
    int c = i / S;
    float v = fmaf((float)p[i], A[c], B[c]);
    p[i] = (_Float16)v;
    o[i] = v;
  }
}

// ---------------------------------------------------------------------------
// conv2 (3x3 VALID, 64->32ch) on normalized a1 + maxpool2 + relu. K=576.
// Weights (32x576 f16 = 36.9KB) are DMA'd into LDS once per workgroup by the
// Tensor Data Mover (TENSOR_LOAD_TO_LDS, 1-D descriptor), then all 8 waves
// read A fragments from LDS.
// ---------------------------------------------------------------------------
__global__ __launch_bounds__(256) void k_conv2(const _Float16* __restrict__ a1h,
    const _Float16* __restrict__ w2h, const float* __restrict__ bias2,
    _Float16* __restrict__ a2h, _Float16* __restrict__ dump) {
  __shared__ _Float16 wsm[18432];                  // [32][576]
  if ((threadIdx.x >> 5) == 0) {                   // wave 0 issues the TDM op
    unsigned long long ga = (unsigned long long)(size_t)w2h;
    unsigned int lds = (unsigned int)(unsigned long long)(size_t)&wsm[0];
    // D# group0: count=1, lds_addr, global_addr[56:0], type=2
    v4u g0 = { 1u, lds, (unsigned int)ga,
               (unsigned int)((ga >> 32) & 0x1ffffffull) | 0x80000000u };
    // D# group1: data_size=2B; tensor_dim0=tile_dim0=18432; dim1=1; stride=18432
    v8i g1 = { 0x00010000, 0x48000000, 0x00010000, 0x48000000,
               0, 18432, 0, 0 };
    v4i g2 = { 0, 0, 0, 0 };
    v4i g3 = { 0, 0, 0, 0 };
    v8i g4 = { 0, 0, 0, 0, 0, 0, 0, 0 };
    __builtin_amdgcn_tensor_load_to_lds(g0, g1, g2, g3, g4, 0);
    __builtin_amdgcn_s_wait_tensorcnt(0);
  }
  __syncthreads();

  int task = blockIdx.x * 8 + (threadIdx.x >> 5);  // 0..12287 = 2048*6
  int lane = threadIdx.x & 31;
  int nl   = lane & 15;
  int nlc  = (nl < 13) ? nl : 0;        // address clamp only
  int pr   = task % 6;
  int n    = task / 6;
  int R0   = 2 * pr;
  const _Float16* ab = a1h + (size_t)n * 14400;

  v8f acc00 = {}, acc01 = {}, acc10 = {}, acc11 = {};
  for (int kc = 0; kc < 18; ++kc) {
    v16h b0, b1v;
    #pragma unroll
    for (int j = 0; j < 16; ++j) {
      int kg = kc * 32 + bfk(lane, j);
      int ci = kg / 9, rr = (kg % 9) / 3, cc = kg % 3;
      const _Float16* p = ab + ci * 225 + (R0 + rr) * 15 + nlc + cc;
      b0[j]  = p[0];
      b1v[j] = p[15];
    }
    v16h a0  = load_afrag(&wsm[nl * 576 + kc * 32], lane);
    v16h a1f = load_afrag(&wsm[(16 + nl) * 576 + kc * 32], lane);
    acc00 = wmma32(a0,  b0,  acc00);
    acc01 = wmma32(a0,  b1v, acc01);
    acc10 = wmma32(a1f, b0,  acc10);
    acc11 = wmma32(a1f, b1v, acc11);
  }

  bool valid = ((nl & 1) == 0) && nl < 12;
  _Float16* op = valid ? (a2h + (size_t)n * 1152 + pr * 6 + (nl >> 1)) : dump;

  #pragma unroll
  for (int mt = 0; mt < 2; ++mt) {
    v8f cr0 = mt ? acc10 : acc00;
    v8f cr1 = mt ? acc11 : acc01;
    #pragma unroll
    for (int r = 0; r < 8; ++r) {
      int mg  = mt * 16 + r + ((lane >= 16) ? 8 : 0);
      float v = fmaxf(cr0[r], cr1[r]) + bias2[mg];
      v = fmaxf(v, swzf<1>(v));
      op[mg * 36] = (_Float16)fmaxf(v, 0.f);
    }
  }
}

// ---------------------------------------------------------------------------
// conv3 (3x3 VALID, 32->32ch) on normalized a2 + relu. K=288, one wave/image.
// ---------------------------------------------------------------------------
__global__ __launch_bounds__(256) void k_conv3(const _Float16* __restrict__ a2h,
    const _Float16* __restrict__ w3h, const float* __restrict__ bias3,
    _Float16* __restrict__ a3h) {
  int n    = blockIdx.x * 8 + (threadIdx.x >> 5);
  int lane = threadIdx.x & 31;
  int nl   = lane & 15;
  int orow = nl >> 2, ocol = nl & 3;
  const _Float16* ab = a2h + (size_t)n * 1152;

  v8f acc0 = {}, acc1 = {};
  for (int kc = 0; kc < 9; ++kc) {
    v16h b;
    #pragma unroll
    for (int j = 0; j < 16; ++j) {
      int kg = kc * 32 + bfk(lane, j);
      int ci = kg / 9, rr = (kg % 9) / 3, cc = kg % 3;
      b[j] = ab[ci * 36 + (orow + rr) * 6 + ocol + cc];
    }
    v16h a0  = load_afrag(w3h + nl * 288 + kc * 32, lane);
    v16h a1f = load_afrag(w3h + (16 + nl) * 288 + kc * 32, lane);
    acc0 = wmma32(a0,  b, acc0);
    acc1 = wmma32(a1f, b, acc1);
  }
  #pragma unroll
  for (int mt = 0; mt < 2; ++mt) {
    v8f cf = mt ? acc1 : acc0;
    #pragma unroll
    for (int r = 0; r < 8; ++r) {
      int mg = mt * 16 + r + ((lane >= 16) ? 8 : 0);
      a3h[(size_t)n * 512 + mg * 16 + nl] =
          (_Float16)fmaxf(cf[r] + bias3[mg], 0.f);
    }
  }
}

// ---------------------------------------------------------------------------
// VQ on normalized a3 (== h): argmin_k |z-c_k|^2 == argmin_k (|c_k|^2 - 2 z.c_k)
// via WMMA; recon + loss. One wave = one image; fully branch-free.
// ---------------------------------------------------------------------------
__global__ __launch_bounds__(256) void k_vq(const _Float16* __restrict__ a3h,
    const _Float16* __restrict__ cbp, const float* __restrict__ ccsq,
    const float* __restrict__ codebook,
    float* __restrict__ out_recon, float* __restrict__ out_loss) {
  int n    = blockIdx.x * 8 + (threadIdx.x >> 5);
  int lane = threadIdx.x & 31;
  int nl   = lane & 15;
  const _Float16* zb = a3h + (size_t)n * 512;

  v16h az;
  #pragma unroll
  for (int j = 0; j < 16; ++j)
    az[j] = zb[afk(lane, j) * 16 + nl];

  float bestd[8]; int besti[8];
  #pragma unroll
  for (int r = 0; r < 8; ++r) { bestd[r] = 3.4e38f; besti[r] = 0; }

  for (int ctile = 0; ctile < 7; ++ctile) {
    int code = ctile * 16 + nl;
    v16h bcb = load_bfrag(cbp + code * 32, lane);   // padded rows zero
    v8f d = {};
    d = wmma32(az, bcb, d);
    float c2 = ccsq[code];                          // padded entries +huge
    #pragma unroll
    for (int r = 0; r < 8; ++r) {
      float dist = c2 - 2.f * d[r];
      if (dist < bestd[r]) { bestd[r] = dist; besti[r] = code; }
    }
  }
  #pragma unroll
  for (int r = 0; r < 8; ++r) {
    float d = bestd[r]; int i = besti[r];
    argmin_step<1>(d, i);
    argmin_step<2>(d, i);
    argmin_step<4>(d, i);
    argmin_step<8>(d, i);
    besti[r] = i;
  }
  // lane's half-group holds rows r+8*(lane>=16); pick idx for row m = nl.
  int sel = besti[0];
  #pragma unroll
  for (int r = 1; r < 8; ++r) sel = ((nl & 7) == r) ? besti[r] : sel;
  int idx = __shfl(sel, (nl < 8) ? nl : (nl + 8), 32);

  int m  = nl;
  int c0 = (lane < 16) ? 0 : 16;                    // channel half per lane
  const float* cbrow = codebook + idx * 32 + c0;
  float lossacc = 0.f;
  #pragma unroll
  for (int c = 0; c < 16; ++c) {
    float qv = cbrow[c];
    float zv = (float)zb[(c0 + c) * 16 + m];
    out_recon[(size_t)n * 512 + (c0 + c) * 16 + m] = qv;
    float dd = qv - zv;
    lossacc = fmaf(dd, dd, lossacc);
  }
  lossacc += swzf<1>(lossacc);
  lossacc += swzf<2>(lossacc);
  lossacc += swzf<4>(lossacc);
  lossacc += swzf<8>(lossacc);
  lossacc += swzf<16>(lossacc);
  if (lane == 0)
    atomicAdd(out_loss, lossacc * (1.25f / (32768.f * 32.f)));
}

// ---------------------------------------------------------------------------
// blended = 0.5*recon + h, coalesced float4 -> f16x4
// ---------------------------------------------------------------------------
__global__ __launch_bounds__(256) void k_blend(const float* __restrict__ recon,
    const float* __restrict__ h, _Float16* __restrict__ blendh) {
  int i = blockIdx.x * 256 + threadIdx.x;           // < 262144 float4s
  v4f r  = ((const v4f*)recon)[i];
  v4f hh = ((const v4f*)h)[i];
  v4h o;
  #pragma unroll
  for (int j = 0; j < 4; ++j) o[j] = (_Float16)fmaf(0.5f, r[j], hh[j]);
  ((v4h*)blendh)[i] = o;
}

// ---------------------------------------------------------------------------
// FC head: z1 = relu(blend @ fc1^T + b1); y = z1 @ fc2^T + b2; log_softmax.
// ---------------------------------------------------------------------------
__global__ __launch_bounds__(256) void k_fc(const _Float16* __restrict__ blendh,
    const _Float16* __restrict__ f1h, const float* __restrict__ fb1,
    const _Float16* __restrict__ f2p, const float* __restrict__ fb2,
    float* __restrict__ out_logp) {
  __shared__ _Float16 z1s[8][1024];
  int wv   = threadIdx.x >> 5;
  int tile = blockIdx.x * 8 + wv;                   // 0..127
  int lane = threadIdx.x & 31;
  int nl   = lane & 15;

  v8f acc[4] = {};
  for (int kc = 0; kc < 16; ++kc) {
    v16h a = load_afrag(blendh + (size_t)(tile * 16 + nl) * 512 + kc * 32, lane);
    #pragma unroll
    for (int nt = 0; nt < 4; ++nt) {
      v16h b = load_bfrag(f1h + (nt * 16 + nl) * 512 + kc * 32, lane);
      acc[nt] = wmma32(a, b, acc[nt]);
    }
  }
  #pragma unroll
  for (int nt = 0; nt < 4; ++nt) {
    #pragma unroll
    for (int r = 0; r < 8; ++r) {
      int m  = r + ((lane >= 16) ? 8 : 0);
      int jj = nt * 16 + nl;
      z1s[wv][m * 64 + jj] = (_Float16)fmaxf(acc[nt][r] + fb1[jj], 0.f);
    }
  }
  __syncthreads();

  v8f acc2 = {};
  #pragma unroll
  for (int kc = 0; kc < 2; ++kc) {
    v16h a = load_afrag(&z1s[wv][nl * 64 + kc * 32], lane);
    v16h b = load_bfrag(f2p + nl * 64 + kc * 32, lane);   // padded rows zero
    acc2 = wmma32(a, b, acc2);
  }
  #pragma unroll
  for (int r = 0; r < 8; ++r) {
    int m   = r + ((lane >= 16) ? 8 : 0);
    float y = acc2[r] + ((nl < 10) ? fb2[nl] : 0.f);
    float ym = (nl < 10) ? y : -3.0e38f;
    ym = fmaxf(ym, swzf<1>(ym));
    ym = fmaxf(ym, swzf<2>(ym));
    ym = fmaxf(ym, swzf<4>(ym));
    ym = fmaxf(ym, swzf<8>(ym));
    float s = (nl < 10) ? __expf(y - ym) : 0.f;
    s += swzf<1>(s);
    s += swzf<2>(s);
    s += swzf<4>(s);
    s += swzf<8>(s);
    if (nl < 10)
      out_logp[(size_t)(tile * 16 + m) * 10 + nl] = y - ym - __logf(s);
  }
}

// ---------------------------------------------------------------------------
extern "C" void kernel_launch(void* const* d_in, const int* in_sizes, int n_in,
                              void* d_out, int out_size, void* d_ws, size_t ws_size,
                              hipStream_t stream) {
  const float* x         = (const float*)d_in[0];
  const float* conv1_w   = (const float*)d_in[1];
  const float* conv1_b   = (const float*)d_in[2];
  const float* bn1_scale = (const float*)d_in[3];
  const float* bn1_bias  = (const float*)d_in[4];
  const float* conv2_w   = (const float*)d_in[5];
  const float* conv2_b   = (const float*)d_in[6];
  const float* bn2_scale = (const float*)d_in[7];
  const float* bn2_bias  = (const float*)d_in[8];
  const float* conv3_w   = (const float*)d_in[9];
  const float* conv3_b   = (const float*)d_in[10];
  const float* bn3_scale = (const float*)d_in[11];
  const float* bn3_bias  = (const float*)d_in[12];
  const float* codebook  = (const float*)d_in[13];
  const float* fc1_w     = (const float*)d_in[14];
  const float* fc1_b     = (const float*)d_in[15];
  const float* fc2_w     = (const float*)d_in[16];
  const float* fc2_b     = (const float*)d_in[17];

  float* out = (float*)d_out;
  char*  ws  = (char*)d_ws;

  _Float16* a1h   = (_Float16*)(ws + OFF_A1);
  _Float16* a2h   = (_Float16*)(ws + OFF_A2);
  _Float16* a3h   = (_Float16*)(ws + OFF_A3);
  _Float16* blend = (_Float16*)(ws + OFF_BL);
  _Float16* w1p   = (_Float16*)(ws + OFF_W1);
  _Float16* w2h   = (_Float16*)(ws + OFF_W2);
  _Float16* w3h   = (_Float16*)(ws + OFF_W3);
  _Float16* cbp   = (_Float16*)(ws + OFF_CB);
  _Float16* f1h   = (_Float16*)(ws + OFF_F1);
  _Float16* f2p   = (_Float16*)(ws + OFF_F2);
  float* bn1A = (float*)(ws + OFF_BN1A);
  float* bn1B = (float*)(ws + OFF_BN1B);
  float* bn2A = (float*)(ws + OFF_BN2A);
  float* bn2B = (float*)(ws + OFF_BN2B);
  float* bn3A = (float*)(ws + OFF_BN3A);
  float* bn3B = (float*)(ws + OFF_BN3B);
  float* ccsq = (float*)(ws + OFF_CC);
  _Float16* dump = (_Float16*)(ws + OFF_DUMP);

  float* out_recon = out;                 // [2048,32,4,4]
  float* out_h     = out + 1048576;       // [2048,32,4,4]
  float* out_loss  = out + 2097152;       // scalar
  float* out_logp  = out + 2097153;       // [2048,10]

  k_prep<<<128, 256, 0, stream>>>(conv1_w, conv2_w, conv3_w, codebook, fc1_w,
                                  fc2_w, w1p, w2h, w3h, cbp, f1h, f2p, ccsq,
                                  out_loss);
  k_conv1<<<7680, 256, 0, stream>>>(x, w1p, conv1_b, a1h, dump);
  k_bnstats<64, 225><<<64, 256, 0, stream>>>(a1h, bn1_scale, bn1_bias, bn1A, bn1B);
  k_bnapply<64, 225><<<2048, 256, 0, stream>>>(a1h, bn1A, bn1B);
  k_conv2<<<1536, 256, 0, stream>>>(a1h, w2h, conv2_b, a2h, dump);
  k_bnstats<32, 36><<<32, 256, 0, stream>>>(a2h, bn2_scale, bn2_bias, bn2A, bn2B);
  k_bnapply<32, 36><<<2048, 256, 0, stream>>>(a2h, bn2A, bn2B);
  k_conv3<<<256, 256, 0, stream>>>(a2h, w3h, conv3_b, a3h);
  k_bnstats<32, 16><<<32, 256, 0, stream>>>(a3h, bn3_scale, bn3_bias, bn3A, bn3B);
  k_bnapply_out<32, 16><<<2048, 256, 0, stream>>>(a3h, bn3A, bn3B, out_h);
  k_vq<<<256, 256, 0, stream>>>(a3h, cbp, ccsq, codebook, out_recon, out_loss);
  k_blend<<<1024, 256, 0, stream>>>(out_recon, out_h, blend);
  k_fc<<<16, 256, 0, stream>>>(blend, f1h, fc1_b, f2p, fc2_b, out_logp);
}
